// HET_CRL_7636451852357
// MI455X (gfx1250) — matrix operand order repaired and checked by reading, not compiled, not run on hardware
//
#include <hip/hip_runtime.h>

typedef __attribute__((ext_vector_type(16))) __bf16 v16bf;
typedef __attribute__((ext_vector_type(8)))  __bf16 v8bf;
typedef __attribute__((ext_vector_type(4)))  __bf16 v4bf;
typedef __attribute__((ext_vector_type(8)))  float  v8f;
typedef __attribute__((ext_vector_type(4)))  unsigned int uint32x4;
typedef __attribute__((ext_vector_type(8)))  int int32x8;
typedef __attribute__((ext_vector_type(4)))  int int32x4;

#define B_TOT 32768
#define DD    1024
#define MM    2048
#define KK    100
#define BT    16
#define NWAVE 8

// dynamic LDS layout (bytes)
#define X_OFF    0          // 16*1024 bf16 = 32768
#define A_OFF    32768      // 16*2048 bf16 = 65536 (also f32 x-tile staging, 64KB)
#define BB_OFF   98304      // 16*2048 bf16 = 65536
#define SCR_OFF  163840     // 16*16 f32   = 1024
#define GV_OFF   164864     // 16 f32
#define SV_OFF   164928     // 16 f32
#define SMEM_BYTES 164992

static __device__ __forceinline__ float fast_sigmoid(float v) {
    // 1/(1+exp(-v)) with v_rcp_f32 instead of IEEE divide
    return __builtin_amdgcn_rcpf(1.0f + __expf(-v));
}

static __device__ __forceinline__ v16bf cat8(v8bf lo, v8bf hi) {
    return __builtin_shufflevector(lo, hi, 0,1,2,3,4,5,6,7,8,9,10,11,12,13,14,15);
}

// out[m, j] = sigmoid( sum_k in[m,k] * W[j,k] + bias[j] ),  m in [0,16), j in [0,2048)
// Two N-tiles per wave iteration: shared A fragment, two independent WMMA chains.
static __device__ __forceinline__ void mlp_layer(
    const __bf16* __restrict__ in_lds, int kdim,
    const __bf16* __restrict__ W, const float* __restrict__ bias,
    __bf16* __restrict__ out_lds)
{
    const int lane = threadIdx.x & 31;
    const int wave = threadIdx.x >> 5;
    const int lo16 = (lane < 16) ? 1 : 0;
    const int nsub = lane & 15;
    const int asel = lo16 ? 0 : 16;   // A lane K-offset (contiguous, matches B)
    const int bsel = lo16 ? 0 : 16;   // B lane K-offset (contiguous, matches A)
    const int moff = lo16 ? 0 : 8;    // C/D row offset

    const __bf16* arow = in_lds + (size_t)nsub * kdim + asel;

    for (int t0 = wave * 2; t0 < MM / 16; t0 += 2 * NWAVE) {
        const int j0 = t0 * 16;
        const int j1 = j0 + 16;
        const __bf16* w0 = W + (size_t)(j0 + nsub) * kdim + bsel;
        const __bf16* w1 = W + (size_t)(j1 + nsub) * kdim + bsel;
        v8f acc0 = {};
        v8f acc1 = {};
        #pragma unroll 4
        for (int k = 0; k < kdim; k += 32) {
            v8bf alo = *(const v8bf*)(arow + k);        // K = asel..asel+7
            v8bf ahi = *(const v8bf*)(arow + k + 8);    // K = asel+8..
            v16bf a = cat8(alo, ahi);
            v8bf b0lo = *(const v8bf*)(w0 + k);
            v8bf b0hi = *(const v8bf*)(w0 + k + 8);
            v8bf b1lo = *(const v8bf*)(w1 + k);
            v8bf b1hi = *(const v8bf*)(w1 + k + 8);
            v16bf b0 = cat8(b0lo, b0hi);
            v16bf b1 = cat8(b1lo, b1hi);
            acc0 = __builtin_amdgcn_wmma_f32_16x16x32_bf16(
                false, a, false, b0, (short)0, acc0, false, false);
            acc1 = __builtin_amdgcn_wmma_f32_16x16x32_bf16(
                false, a, false, b1, (short)0, acc1, false, false);
        }
        const float bc0 = bias[j0 + nsub];
        const float bc1 = bias[j1 + nsub];
        #pragma unroll
        for (int r = 0; r < 8; ++r) {
            out_lds[(size_t)(r + moff) * MM + (j0 + nsub)] = (__bf16)fast_sigmoid(acc0[r] + bc0);
            out_lds[(size_t)(r + moff) * MM + (j1 + nsub)] = (__bf16)fast_sigmoid(acc1[r] + bc1);
        }
    }
}

// outv[row] = dot(act[row, :], w) + bias   (row in [0,16))
static __device__ __forceinline__ void head_dot(
    const __bf16* __restrict__ act, const float* __restrict__ w,
    float bias, float* __restrict__ scr, float* __restrict__ outv)
{
    const int row = threadIdx.x >> 4;   // 0..15
    const int seg = threadIdx.x & 15;   // 0..15
    const __bf16* p  = act + (size_t)row * MM + seg * 128;
    const float*  wp = w + seg * 128;
    float acc = 0.f;
    for (int c = 0; c < 128; c += 8) {
        v8bf v = *(const v8bf*)(p + c);
        #pragma unroll
        for (int u = 0; u < 8; ++u) acc += (float)v[u] * wp[c + u];
    }
    scr[row * 16 + seg] = acc;
    __syncthreads();
    if (seg == 0) {
        float s = 0.f;
        #pragma unroll
        for (int i = 0; i < 16; ++i) s += scr[row * 16 + i];
        outv[row] = s + bias;
    }
    __syncthreads();
}

__global__ void __launch_bounds__(256)
het_fused_mlp_kernel(
    const float* __restrict__ x,
    const __bf16* __restrict__ g1, const float* __restrict__ g1b,
    const __bf16* __restrict__ g2, const float* __restrict__ g2b,
    const __bf16* __restrict__ g3, const float* __restrict__ g3b,
    const float* __restrict__ g4w, const float* __restrict__ g4b,
    const __bf16* __restrict__ s1, const float* __restrict__ s1b,
    const __bf16* __restrict__ s2, const float* __restrict__ s2b,
    const __bf16* __restrict__ s3, const float* __restrict__ s3b,
    const float* __restrict__ s4w, const float* __restrict__ s4b,
    const float* __restrict__ tmp_bi,
    float* __restrict__ out)
{
    extern __shared__ char smem[];
    __bf16* xbuf = (__bf16*)(smem + X_OFF);
    __bf16* actA = (__bf16*)(smem + A_OFF);
    __bf16* actB = (__bf16*)(smem + BB_OFF);
    float*  xstg = (float*)(smem + A_OFF);    // f32 staging for TDM x-tile (64KB)
    float*  scr  = (float*)(smem + SCR_OFF);
    float*  gv   = (float*)(smem + GV_OFF);
    float*  sv   = (float*)(smem + SV_OFF);

    const int tile = blockIdx.x;

    // Phase 0a: Tensor Data Mover pulls x tile [16,1024] f32 into LDS staging.
    // 2D D#: data_size=4B, tensor 1024 x 32768 (dim0 x dim1), tile 1024 x 16,
    // global_addr points at the tile start (row tile*16), stride0 = 1024.
    if ((threadIdx.x >> 5) == 0) {
        const unsigned long long ga =
            (unsigned long long)(const void*)(x + (size_t)tile * BT * DD);
        uint32x4 g0;
        g0[0] = 1u;                                   // count=1, user descriptor
        g0[1] = (unsigned)A_OFF;                      // lds_addr (bytes)
        g0[2] = (unsigned)(ga & 0xFFFFFFFFu);         // global_addr[31:0]
        g0[3] = (unsigned)((ga >> 32) & 0x01FFFFFFu)  // global_addr[56:32]
              | (2u << 30);                           // type=2 ("image")
        int32x8 g1d;
        g1d[0] = 0x00020000;                          // data_size=2 -> 4 bytes
        g1d[1] = (int)(((unsigned)DD & 0xFFFFu) << 16);   // tensor_dim0 lo16
        g1d[2] = (int)0x80000000u;                    // tensor_dim1 lo16 = 32768
        g1d[3] = (int)(((unsigned)DD & 0xFFFFu) << 16);   // tile_dim0 = 1024
        g1d[4] = BT;                                  // tile_dim1 = 16
        g1d[5] = DD;                                  // tensor_dim0_stride lo32
        g1d[6] = 0;
        g1d[7] = 0;
        int32x4 gz4 = {};
        int32x8 gz8 = {};
        __builtin_amdgcn_tensor_load_to_lds(g0, g1d, gz4, gz4, gz8, 0);
        __builtin_amdgcn_s_wait_tensorcnt(0);
    }
    __syncthreads();

    // Phase 0b: convert staged f32 -> bf16 xbuf
    for (int i = threadIdx.x; i < BT * DD / 4; i += 256) {
        float4 v = ((const float4*)xstg)[i];
        v4bf o = { (__bf16)v.x, (__bf16)v.y, (__bf16)v.z, (__bf16)v.w };
        *(v4bf*)(xbuf + i * 4) = o;
    }
    __syncthreads();

    // ---- g branch ----
    mlp_layer(xbuf, DD, g1, g1b, actA); __syncthreads();
    mlp_layer(actA, MM, g2, g2b, actB); __syncthreads();
    mlp_layer(actB, MM, g3, g3b, actA); __syncthreads();
    head_dot(actA, g4w, g4b[0], scr, gv);

    // ---- s branch (x tile still resident) ----
    mlp_layer(xbuf, DD, s1, s1b, actA); __syncthreads();
    mlp_layer(actA, MM, s2, s2b, actB); __syncthreads();
    mlp_layer(actB, MM, s3, s3b, actA); __syncthreads();
    head_dot(actA, s4w, s4b[0], scr, sv);

    // ---- epilogue: pro[row, k] ----
    if (threadIdx.x < BT) {
        const int row = threadIdx.x;
        const float g = gv[row];
        const float s = 1.0f + __expf(sv[row] * 0.1f);
        const float inv_s = __builtin_amdgcn_rcpf(s);
        float cum = 0.0f;
        float* orow = out + ((size_t)tile * BT + row) * KK;
        for (int k = 0; k < KK; ++k) {
            float bik = (k == 0) ? 0.0f : tmp_bi[k - 1];
            float t = (k == KK - 1) ? 1.0e6f : (bik - g) * inv_s;
            float sig = __builtin_amdgcn_rcpf(1.0f + __expf(-t));
            orow[k] = sig * __expf(cum);                 // sigmoid(t) * exp(excl cumsum)
            float sp = fmaxf(t, 0.0f) + log1pf(__expf(-fabsf(t)));  // softplus(t)
            cum -= sp;                                   // cum += log_sigmoid(-t)
        }
    }
}

__global__ void __launch_bounds__(256)
het_cvt_bf16_kernel(const float* __restrict__ src, __bf16* __restrict__ dst, int n4)
{
    int i = blockIdx.x * 256 + threadIdx.x;
    if (i < n4) {
        float4 v = ((const float4*)src)[i];
        v4bf o = { (__bf16)v.x, (__bf16)v.y, (__bf16)v.z, (__bf16)v.w };
        ((v4bf*)dst)[i] = o;
    }
}

extern "C" void kernel_launch(void* const* d_in, const int* in_sizes, int n_in,
                              void* d_out, int out_size, void* d_ws, size_t ws_size,
                              hipStream_t stream) {
    (void)in_sizes; (void)n_in; (void)out_size; (void)ws_size;
    const float* x    = (const float*)d_in[0];
    const float* g1wf = (const float*)d_in[1];
    const float* g1b  = (const float*)d_in[2];
    const float* g2wf = (const float*)d_in[3];
    const float* g2b  = (const float*)d_in[4];
    const float* g3wf = (const float*)d_in[5];
    const float* g3b  = (const float*)d_in[6];
    const float* g4w  = (const float*)d_in[7];
    const float* g4b  = (const float*)d_in[8];
    const float* s1wf = (const float*)d_in[9];
    const float* s1b  = (const float*)d_in[10];
    const float* s2wf = (const float*)d_in[11];
    const float* s2b  = (const float*)d_in[12];
    const float* s3wf = (const float*)d_in[13];
    const float* s3b  = (const float*)d_in[14];
    const float* s4w  = (const float*)d_in[15];
    const float* s4b  = (const float*)d_in[16];
    const float* tbi  = (const float*)d_in[17];
    float* out = (float*)d_out;

    // bf16 weight cache in workspace (element offsets)
    __bf16* ws = (__bf16*)d_ws;
    __bf16* g1 = ws + 0;          // 2097152 (M*D)
    __bf16* g2 = ws + 2097152;    // 4194304 (M*M)
    __bf16* g3 = ws + 6291456;    // 4194304
    __bf16* s1 = ws + 10485760;   // 2097152
    __bf16* s2 = ws + 12582912;   // 4194304
    __bf16* s3 = ws + 16777216;   // 4194304  -> total 41.9 MB

    struct Cv { const float* src; __bf16* dst; int n; } cv[6] = {
        { g1wf, g1, MM * DD }, { g2wf, g2, MM * MM }, { g3wf, g3, MM * MM },
        { s1wf, s1, MM * DD }, { s2wf, s2, MM * MM }, { s3wf, s3, MM * MM },
    };
    for (int i = 0; i < 6; ++i) {
        int n4 = cv[i].n / 4;
        het_cvt_bf16_kernel<<<(n4 + 255) / 256, 256, 0, stream>>>(cv[i].src, cv[i].dst, n4);
    }

    het_fused_mlp_kernel<<<B_TOT / BT, 256, SMEM_BYTES, stream>>>(
        x, g1, g1b, g2, g2b, g3, g3b, g4w, g4b,
        s1, s1b, s2, s2b, s3, s3b, s4w, s4b, tbi, out);
}
